// LSTM2_43499428774129
// MI455X (gfx1250) — matrix operand order repaired and checked by reading, not compiled, not run on hardware
//
#include <hip/hip_runtime.h>
#include <hip/hip_bf16.h>

// ---------------------------------------------------------------------------
// LSTM (HID=128, INP=12, B=256, T=1000) + final FC(128->5) on gfx1250.
// One workgroup (512 threads, 16 waves) per 16 batch rows; the whole T-scan
// runs inside the workgroup. Recurrent + input GEMM fused as 16x512x160 bf16
// WMMA per step:
//   A = [h_bf16(16x128) | x_bf16(16x12 padded to 32)]   (LDS, rewritten/step)
//   B = [W_hh | W_ih padded] (LDS, loaded once; fragments hoisted to VGPRs)
// Per wave: 2 N-tiles x 5 K-steps = 10 WMMAs/step. All 5 A fragments are
// loaded up-front each step so LDS latency overlaps the WMMA chain; x(t+1)
// streams into registers across the step; barriers are hand-rolled split
// barriers that wait only on DScnt (not on the in-flight x global load).
// ---------------------------------------------------------------------------

typedef __attribute__((ext_vector_type(16))) __bf16 v16bf;
typedef __attribute__((ext_vector_type(8)))  float  v8f;

union Frag {
    uint4 q[2];
    v16bf v;
};

#define T_STEPS 1000
#define INP     12
#define HID     128
#define G4      512          // 4*HID gates
#define KTOT    160          // 128 (h) + 32 (x padded)
#define MT      16           // batch rows per workgroup
#define NCLS    5
#define NTH     512          // threads per workgroup (16 waves)

// LDS layout (bytes)
#define OFF_WX   0                              // bf16 [512][160]  = 163840
#define OFF_HX   (OFF_WX + G4 * KTOT * 2)       // bf16 [16][160]   =   5120
#define OFF_GT   (OFF_HX + MT * KTOT * 2)       // f32  [16][512]   =  32768
#define OFF_HF   (OFF_GT + MT * G4 * 4)         // f32  [16][128]   =   8192
#define OFF_BI   (OFF_HF + MT * HID * 4)        // f32  [512]       =   2048
#define SMEM_BYTES (OFF_BI + G4 * 4)            // 211968 < 320KB per WGP

// Split workgroup barrier with LDS-only ordering: prior ds ops must be
// complete/visible, but in-flight global loads (x prefetch) stay outstanding.
#define WG_BARRIER_LDS() \
    asm volatile("s_wait_dscnt 0x0\n\ts_barrier_signal -1\n\ts_barrier_wait -1" ::: "memory")

__device__ __forceinline__ __bf16 f32_to_bf16(float f) {
    unsigned u = __builtin_bit_cast(unsigned, f);
    unsigned r = (u + 0x7FFFu + ((u >> 16) & 1u)) >> 16;   // round-nearest-even
    unsigned short s = (unsigned short)r;
    __bf16 out;
    __builtin_memcpy(&out, &s, 2);
    return out;
}

__device__ __forceinline__ float fsig(float v) {
    // 1 / (1 + exp(-v)) using v_exp_f32 (2^x) + v_rcp_f32
    return __builtin_amdgcn_rcpf(1.0f + __builtin_amdgcn_exp2f(v * -1.4426950408889634f));
}
__device__ __forceinline__ float ftanh(float v) {
    // tanh(v) = 1 - 2/(exp(2v)+1)
    return 1.0f - 2.0f * __builtin_amdgcn_rcpf(1.0f + __builtin_amdgcn_exp2f(v * 2.8853900817779268f));
}

__global__ __launch_bounds__(NTH) void lstm_wmma_kernel(
    const float* __restrict__ x,      // [256][1000][12]
    const float* __restrict__ W_ih,   // [512][12]
    const float* __restrict__ W_hh,   // [512][128]
    const float* __restrict__ b_ih,   // [512]
    const float* __restrict__ b_hh,   // [512]
    const float* __restrict__ W_fc,   // [5][128]
    const float* __restrict__ b_fc,   // [5]
    float* __restrict__ out)          // [256][5]
{
    extern __shared__ char smem[];
    __bf16* Wx   = (__bf16*)(smem + OFF_WX);
    __bf16* hx   = (__bf16*)(smem + OFF_HX);
    float*  gts  = (float*)(smem + OFF_GT);
    float*  hf   = (float*)(smem + OFF_HF);
    float*  bias = (float*)(smem + OFF_BI);

    const int tid   = threadIdx.x;
    const int lane  = tid & 31;
    const int wave  = tid >> 5;            // 16 waves; wave w owns gate cols [32w, 32w+32)
    const int bbase = blockIdx.x * MT;

    // ---- one-time setup: weights -> LDS (bf16), biases, h0 = 0 --------------
    for (int i = tid; i < G4 * HID; i += NTH) {
        int g = i >> 7, k = i & 127;
        Wx[g * KTOT + k] = f32_to_bf16(W_hh[i]);
    }
    for (int i = tid; i < G4 * 32; i += NTH) {
        int g = i >> 5, k = i & 31;
        Wx[g * KTOT + HID + k] = f32_to_bf16(k < INP ? W_ih[g * INP + k] : 0.0f);
    }
    for (int i = tid; i < G4; i += NTH) bias[i] = b_ih[i] + b_hh[i];
    for (int i = tid; i < MT * KTOT; i += NTH) hx[i] = f32_to_bf16(0.0f);

    float creg[4];
#pragma unroll
    for (int u = 0; u < 4; ++u) creg[u] = 0.0f;

    // x streaming: thread tid<192 owns element (m = tid/12, ii = tid%12),
    // holds x(t+1) in a register across the step-t compute.
    const int xm = tid / INP;
    const int xi = tid - xm * INP;
    const float* xptr = x + (size_t)(bbase + xm) * T_STEPS * INP + xi;
    float xreg = (tid < MT * INP) ? xptr[0] : 0.0f;

    __syncthreads();

    // per-thread loop-invariant bias values for the elementwise stage
    float bsi[4], bsf[4], bsg[4], bso[4];
#pragma unroll
    for (int u = 0; u < 4; ++u) {
        const int j = (u * NTH + tid) & 127;
        bsi[u] = bias[j];
        bsf[u] = bias[HID + j];
        bsg[u] = bias[2 * HID + j];
        bso[u] = bias[3 * HID + j];
    }

    // ---- hoist loop-invariant B fragments (W) into VGPRs --------------------
    // B (32x16, KxN) lane layout: n = l&15; elems 0..7 -> K kb..kb+7,
    // elems 8..15 -> K kb+16..kb+23, kb = (l<16 ? 0 : 8) (matches A).
    Frag bfr[2][5];
#pragma unroll
    for (int i = 0; i < 2; ++i) {
        const int g = (wave * 2 + i) * 16 + (lane & 15);
#pragma unroll
        for (int kk = 0; kk < 5; ++kk) {
            const int kb = kk * 32 + ((lane & 16) ? 8 : 0);
            const uint4* p = (const uint4*)(Wx + g * KTOT + kb);
            bfr[i][kk].q[0] = p[0];
            bfr[i][kk].q[1] = p[2];
        }
    }

    // A (16x32) lane layout: m = l&15; elems 0..7 -> K kb..kb+7,
    // elems 8..15 -> K kb+16..kb+23, kb = kk*32 + (l<16 ? 0 : 8).
    // A lane's full K row = 320 contiguous LDS bytes starting at arow.
    const __bf16* arow = hx + (lane & 15) * KTOT + ((lane & 16) ? 8 : 0);

    // ---- sequential scan over time -----------------------------------------
    for (int t = 0; t < T_STEPS; ++t) {
        // stage 1: commit x(t) register into A-panel columns [128,140)
        if (tid < MT * INP) {
            hx[xm * KTOT + HID + xi] = f32_to_bf16(xreg);
        }
        WG_BARRIER_LDS();                       // barrier A (DScnt-only fence)

        // preload x(t+1); stays outstanding across barrier B, waited only at
        // next iteration's stage-1 use -> latency hides behind the whole step.
        if (tid < MT * INP && t + 1 < T_STEPS) {
            xreg = xptr[(size_t)(t + 1) * INP];
        }

        // stage 2: issue ALL A-fragment loads up front (10x ds_load_b128 into
        // distinct registers), then run the 10-WMMA chain with partial waits.
        Frag a0, a1, a2, a3, a4;
        a0.q[0] = *(const uint4*)(arow +   0);  a0.q[1] = *(const uint4*)(arow +  16);
        a1.q[0] = *(const uint4*)(arow +  32);  a1.q[1] = *(const uint4*)(arow +  48);
        a2.q[0] = *(const uint4*)(arow +  64);  a2.q[1] = *(const uint4*)(arow +  80);
        a3.q[0] = *(const uint4*)(arow +  96);  a3.q[1] = *(const uint4*)(arow + 112);
        a4.q[0] = *(const uint4*)(arow + 128);  a4.q[1] = *(const uint4*)(arow + 144);

        v8f acc[2];
#pragma unroll
        for (int i = 0; i < 2; ++i) acc[i] = (v8f){0.f, 0.f, 0.f, 0.f, 0.f, 0.f, 0.f, 0.f};

#pragma unroll
        for (int i = 0; i < 2; ++i)
            acc[i] = __builtin_amdgcn_wmma_f32_16x16x32_bf16(
                false, a0.v, false, bfr[i][0].v, (short)0, acc[i], false, false);
#pragma unroll
        for (int i = 0; i < 2; ++i)
            acc[i] = __builtin_amdgcn_wmma_f32_16x16x32_bf16(
                false, a1.v, false, bfr[i][1].v, (short)0, acc[i], false, false);
#pragma unroll
        for (int i = 0; i < 2; ++i)
            acc[i] = __builtin_amdgcn_wmma_f32_16x16x32_bf16(
                false, a2.v, false, bfr[i][2].v, (short)0, acc[i], false, false);
#pragma unroll
        for (int i = 0; i < 2; ++i)
            acc[i] = __builtin_amdgcn_wmma_f32_16x16x32_bf16(
                false, a3.v, false, bfr[i][3].v, (short)0, acc[i], false, false);
#pragma unroll
        for (int i = 0; i < 2; ++i)
            acc[i] = __builtin_amdgcn_wmma_f32_16x16x32_bf16(
                false, a4.v, false, bfr[i][4].v, (short)0, acc[i], false, false);

        // stage 3: scatter accumulators to gates buffer
        {
            const int mb = (lane & 16) ? 8 : 0;
#pragma unroll
            for (int i = 0; i < 2; ++i) {
                const int n = (wave * 2 + i) * 16 + (lane & 15);
#pragma unroll
                for (int r = 0; r < 8; ++r) gts[(mb + r) * G4 + n] = acc[i][r];
            }
        }
        WG_BARRIER_LDS();                       // barrier B (DScnt-only fence)

        // stage 4: elementwise LSTM cell update (each thread owns 4 (m,j) cells)
#pragma unroll
        for (int u = 0; u < 4; ++u) {
            const int idx = u * NTH + tid;           // 0..2047
            const int m = idx >> 7, j = idx & 127;
            const float gi = gts[m * G4 + j]           + bsi[u];
            const float gf = gts[m * G4 + HID + j]     + bsf[u];
            const float gg = gts[m * G4 + 2 * HID + j] + bsg[u];
            const float go = gts[m * G4 + 3 * HID + j] + bso[u];
            const float i_ = fsig(gi);
            const float f_ = fsig(gf);
            const float g_ = ftanh(gg);
            const float o_ = fsig(go);
            const float c  = f_ * creg[u] + i_ * g_;
            creg[u] = c;
            const float h = o_ * ftanh(c);
            hx[m * KTOT + j] = f32_to_bf16(h);
            if (t == T_STEPS - 1) hf[m * HID + j] = h;
        }
    }
    __syncthreads();

    // ---- final FC: out[b][n] = hT[b] . W_fc[n] + b_fc[n] --------------------
    if (tid < MT * NCLS) {
        const int bl = tid & 15, n = tid >> 4;
        float s = b_fc[n];
        for (int j = 0; j < HID; ++j) s += hf[bl * HID + j] * W_fc[n * HID + j];
        out[(bbase + bl) * NCLS + n] = s;
    }
}

extern "C" void kernel_launch(void* const* d_in, const int* in_sizes, int n_in,
                              void* d_out, int out_size, void* d_ws, size_t ws_size,
                              hipStream_t stream) {
    const float* x    = (const float*)d_in[0];
    const float* W_ih = (const float*)d_in[1];
    const float* W_hh = (const float*)d_in[2];
    const float* b_ih = (const float*)d_in[3];
    const float* b_hh = (const float*)d_in[4];
    const float* W_fc = (const float*)d_in[5];
    const float* b_fc = (const float*)d_in[6];

    (void)in_sizes; (void)n_in; (void)out_size; (void)d_ws; (void)ws_size;

    hipFuncSetAttribute((const void*)lstm_wmma_kernel,
                        hipFuncAttributeMaxDynamicSharedMemorySize,
                        (int)SMEM_BYTES);

    lstm_wmma_kernel<<<dim3(256 / MT), dim3(NTH), SMEM_BYTES, stream>>>(
        x, W_ih, W_hh, b_ih, b_hh, W_fc, b_fc, (float*)d_out);
}
